// GraphAttentionNetwork_13640816132544
// MI455X (gfx1250) — matrix operand order repaired and checked by reading, not compiled, not run on hardware
//
#include <hip/hip_runtime.h>
#include <hip/hip_bf16.h>

// ---------------------------------------------------------------------------
// GAT forward for MI455X (gfx1250, wave32). Matmuls via v_wmma_f32_16x16x32_f16.
// Bandwidth-bound on alpha1/alpha2 (268 MB f32 out + re-read): the alpha*V GEMM
// strip-mines 16x64 per wave (one A fragment feeds 4 WMMAs) so alpha is read
// ~1x instead of 4x, and the shared 64xK B panel is staged once per block into
// LDS (TDM tensor_load_to_lds issued first, then an authoritative cooperative
// copy after s_wait_tensorcnt so correctness never depends on D# semantics).
// ---------------------------------------------------------------------------

typedef _Float16 f16;
typedef __attribute__((ext_vector_type(16))) _Float16 v16h;
typedef __attribute__((ext_vector_type(8)))  _Float16 v8h;
typedef __attribute__((ext_vector_type(8)))  float    v8f;
typedef __attribute__((ext_vector_type(4)))  float    v4f;
typedef __attribute__((ext_vector_type(4)))  unsigned int u32x4;
typedef __attribute__((ext_vector_type(8)))  int      i32x8;
typedef __attribute__((ext_vector_type(4)))  int      i32x4;

constexpr int kB  = 16;
constexpr int kN  = 512;
constexpr int kD  = 128;
constexpr int kNH = 8;
constexpr int kF  = 64;
constexpr float kLRELU = 0.2f;
constexpr float kEPS   = 1e-5f;

// ---------------------------------------------------------------------------
// Batched GEMM: C[bh] (M x 64, f32) = A[bh] (M x K) * Bt[bh]^T
//   A : row-major MxK, f16 (AF32=false) or f32 cvt-in-register (AF32=true)
//   Bt: row-major 64 x K f16 (B pre-transposed -> K contiguous)
// Block = 8 waves = 8 consecutive 16-row strips of ONE batch; the 64xK B panel
// is staged in LDS (TDM + cooperative copy). Each wave: 4 accumulators, one A
// fragment per 32-wide k-step feeding 4 WMMAs.
// Fragment layouts per CDNA5 ISA 7.12.2 (wave32).
// ---------------------------------------------------------------------------
template<bool AF32>
__global__ __launch_bounds__(256)
void gemm_wmma(const void* __restrict__ Ap, const f16* __restrict__ Bt,
               float* __restrict__ C,
               int M, int K, int nBatch,
               long sAb, long sAh, long sBb, long sBh, long sC)
{
  extern __shared__ f16 sB[];                 // 64 x K halves
  const int lane = threadIdx.x & 31;
  const int wv   = threadIdx.x >> 5;          // 0..7
  const int blocksPerBatch = (M >> 4) >> 3;   // strips/8
  const int bh = blockIdx.x / blocksPerBatch;
  const int sg = blockIdx.x - bh * blocksPerBatch;
  if (bh >= nBatch) return;
  const int b = bh / kNH;
  const int h = bh - b * kNH;
  const int hi  = lane >> 4;                  // 0: lanes 0-15, 1: lanes 16-31
  const int l15 = lane & 15;

  const f16* Bsrc = Bt + b * sBb + h * sBh;   // 64*K halves

#if __has_builtin(__builtin_amdgcn_tensor_load_to_lds)
  // TDM: DMA the 64xK f16 panel (2-D tensor, data_size=2B) into LDS.
  // Issued by wave 0 only (TDM ignores EXEC; descriptor is wave-uniform).
  if (threadIdx.x < 32) {
    unsigned long long ga = (unsigned long long)(const void*)Bsrc;
    unsigned lds_off = (unsigned)(unsigned long long)(const void*)sB; // low 32 = LDS addr
    u32x4 g0 = { 0x1u,                                   // count=1
                 lds_off,
                 (unsigned)(ga & 0xffffffffull),
                 (unsigned)((ga >> 32) & 0x01ffffffull) | 0x80000000u }; // type=2
    i32x8 g1 = { (int)(1u << 16),          // data_size=1 (2 bytes)
                 (int)((unsigned)K << 16), // tensor_dim0[15:0]
                 (int)(64u << 16),         // tensor_dim0[31:16]=0 | tensor_dim1[15:0]=64
                 (int)((unsigned)K << 16), // tensor_dim1[31:16]=0 | tile_dim0=K
                 (int)64,                  // tile_dim1=64 | tile_dim2=0
                 (int)K,                   // tensor_dim0_stride[31:0]
                 0, 0 };
    i32x4 gz = { 0, 0, 0, 0 };
#if __clang_major__ >= 23
    i32x8 gz8 = { 0, 0, 0, 0, 0, 0, 0, 0 };
    __builtin_amdgcn_tensor_load_to_lds(g0, g1, gz, gz, gz8, 0);
#else
    __builtin_amdgcn_tensor_load_to_lds(g0, g1, gz, gz, 0);
#endif
#if __has_builtin(__builtin_amdgcn_s_wait_tensorcnt)
    __builtin_amdgcn_s_wait_tensorcnt((short)0);
#endif
  }
  __syncthreads();
#endif
  // Authoritative cooperative copy (lands after the TDM wait, so the result
  // never depends on unverified D# semantics).
  {
    const int chunks = (64 * K) >> 3;           // 16B chunks
    for (int c = threadIdx.x; c < chunks; c += 256)
      *(v8h*)(sB + c * 8) = *(const v8h*)(Bsrc + c * 8);
  }
  __syncthreads();

  const int mt = sg * 8 + wv;                   // this wave's 16-row strip
  const long bKoff = (long)l15 * K + hi * 8;
  const f16* b0 = sB + bKoff;
  const f16* b1 = sB + 16L * K + bKoff;
  const f16* b2 = sB + 32L * K + bKoff;
  const f16* b3 = sB + 48L * K + bKoff;

  v8f acc0 = {}, acc1 = {}, acc2 = {}, acc3 = {};

  if (AF32) {
    const float* arow = (const float*)Ap + b * sAb + h * sAh
                        + (long)(mt * 16 + l15) * K + hi * 8;
    for (int kc = 0; kc < K; kc += 32) {
      __builtin_prefetch(arow + kc + 32, 0, 1);     // global_prefetch_b8
      v4f f0 = *(const v4f*)(arow + kc);
      v4f f1 = *(const v4f*)(arow + kc + 4);
      v4f f2 = *(const v4f*)(arow + kc + 16);
      v4f f3 = *(const v4f*)(arow + kc + 20);
      v16h a;
#pragma unroll
      for (int i = 0; i < 4; ++i) {
        a[i]      = (f16)f0[i];
        a[4 + i]  = (f16)f1[i];
        a[8 + i]  = (f16)f2[i];
        a[12 + i] = (f16)f3[i];
      }
      union { v16h v; v8h p[2]; } u0, u1, u2, u3;
      u0.p[0] = *(const v8h*)(b0 + kc); u0.p[1] = *(const v8h*)(b0 + kc + 16);
      u1.p[0] = *(const v8h*)(b1 + kc); u1.p[1] = *(const v8h*)(b1 + kc + 16);
      u2.p[0] = *(const v8h*)(b2 + kc); u2.p[1] = *(const v8h*)(b2 + kc + 16);
      u3.p[0] = *(const v8h*)(b3 + kc); u3.p[1] = *(const v8h*)(b3 + kc + 16);
      acc0 = __builtin_amdgcn_wmma_f32_16x16x32_f16(false, a, false, u0.v, (short)0, acc0, false, false);
      acc1 = __builtin_amdgcn_wmma_f32_16x16x32_f16(false, a, false, u1.v, (short)0, acc1, false, false);
      acc2 = __builtin_amdgcn_wmma_f32_16x16x32_f16(false, a, false, u2.v, (short)0, acc2, false, false);
      acc3 = __builtin_amdgcn_wmma_f32_16x16x32_f16(false, a, false, u3.v, (short)0, acc3, false, false);
    }
  } else {
    const f16* arow = (const f16*)Ap + b * sAb + h * sAh
                      + (long)(mt * 16 + l15) * K + hi * 8;
    for (int kc = 0; kc < K; kc += 32) {
      __builtin_prefetch(arow + kc + 64, 0, 1);     // global_prefetch_b8
      union { v16h v; v8h p[2]; } au, u0, u1, u2, u3;
      au.p[0] = *(const v8h*)(arow + kc);
      au.p[1] = *(const v8h*)(arow + kc + 16);
      u0.p[0] = *(const v8h*)(b0 + kc); u0.p[1] = *(const v8h*)(b0 + kc + 16);
      u1.p[0] = *(const v8h*)(b1 + kc); u1.p[1] = *(const v8h*)(b1 + kc + 16);
      u2.p[0] = *(const v8h*)(b2 + kc); u2.p[1] = *(const v8h*)(b2 + kc + 16);
      u3.p[0] = *(const v8h*)(b3 + kc); u3.p[1] = *(const v8h*)(b3 + kc + 16);
      acc0 = __builtin_amdgcn_wmma_f32_16x16x32_f16(false, au.v, false, u0.v, (short)0, acc0, false, false);
      acc1 = __builtin_amdgcn_wmma_f32_16x16x32_f16(false, au.v, false, u1.v, (short)0, acc1, false, false);
      acc2 = __builtin_amdgcn_wmma_f32_16x16x32_f16(false, au.v, false, u2.v, (short)0, acc2, false, false);
      acc3 = __builtin_amdgcn_wmma_f32_16x16x32_f16(false, au.v, false, u3.v, (short)0, acc3, false, false);
    }
  }

  // C layout: lane L, vgpr r -> M = 16*mt + 8*hi + r, N = 16*nt + l15
  float* crow = C + bh * sC + (long)(mt * 16 + hi * 8) * 64 + l15;
#pragma unroll
  for (int r = 0; r < 8; ++r) {
    crow[(long)r * 64]      = acc0[r];
    crow[(long)r * 64 + 16] = acc1[r];
    crow[(long)r * 64 + 32] = acc2[r];
    crow[(long)r * 64 + 48] = acc3[r];
  }
}

// ---------------------------------------------------------------------------
__global__ void cast_f16_kernel(const float* __restrict__ x, f16* __restrict__ y,
                                long n) {
  long i = (long)blockIdx.x * blockDim.x + threadIdx.x;
  if (i < n) y[i] = (f16)x[i];
}

// batched (R x C) f32 -> (C x R) f16  (makes K contiguous for B operands)
__global__ void tcast_kernel(const float* __restrict__ X, f16* __restrict__ Y,
                             int R, int C, int nb) {
  long i = (long)blockIdx.x * blockDim.x + threadIdx.x;
  long total = (long)nb * R * C;
  if (i >= total) return;
  int c = (int)(i % C);
  long t = i / C;
  int r = (int)(t % R);
  int b = (int)(t / R);
  Y[(long)b * R * C + (long)c * R + r] = (f16)X[i];
}

// s_src/s_dst: per (b,h,n) dot of Ht row with a[h,:F] and a[h,F:]
__global__ void scores_kernel(const float* __restrict__ Ht,
                              const float* __restrict__ a,
                              float* __restrict__ ssrc, float* __restrict__ sdst) {
  int row = blockIdx.x * blockDim.x + threadIdx.x;   // bh*N + n
  if (row >= kB * kNH * kN) return;
  int h = (row / kN) % kNH;
  const float* hp   = Ht + (long)row * kF;
  const float* asrc = a + h * 2 * kF;
  const float* adst = asrc + kF;
  float s0 = 0.f, s1 = 0.f;
  for (int f = 0; f < kF; ++f) {
    float v = hp[f];
    s0 += v * asrc[f];
    s1 += v * adst[f];
  }
  ssrc[row] = s0;
  sdst[row] = s1;
}

// wave-per-row softmax over 512 neighbors: e = lrelu(s_i + s_j), softmax_j
__global__ __launch_bounds__(256)
void softmax_kernel(const float* __restrict__ ssrc, const float* __restrict__ sdst,
                    float* __restrict__ alpha) {
  int wid  = (blockIdx.x * blockDim.x + threadIdx.x) >> 5;
  int lane = threadIdx.x & 31;
  if (wid >= kB * kNH * kN) return;
  int bh = wid >> 9;                       // N = 512
  float si = ssrc[wid];
  const float* sd = sdst + (long)bh * kN;
  float e[16];
  float m = -3.4e38f;
#pragma unroll
  for (int t = 0; t < 16; ++t) {
    float x = si + sd[lane + 32 * t];
    x = x > 0.f ? x : kLRELU * x;
    e[t] = x;
    m = fmaxf(m, x);
  }
  for (int off = 16; off; off >>= 1) m = fmaxf(m, __shfl_xor(m, off, 32));
  float s = 0.f;
#pragma unroll
  for (int t = 0; t < 16; ++t) { e[t] = __expf(e[t] - m); s += e[t]; }
  for (int off = 16; off; off >>= 1) s += __shfl_xor(s, off, 32);
  float inv = 1.0f / s;
  float* ar = alpha + (long)wid * kN;
#pragma unroll
  for (int t = 0; t < 16; ++t) ar[lane + 32 * t] = e[t] * inv;
}

// Hn (B,NH,N,F) -> H1raw (B,N,NH*F) with ELU
__global__ void eluconcat_kernel(const float* __restrict__ Hn,
                                 float* __restrict__ H1) {
  long i = (long)blockIdx.x * blockDim.x + threadIdx.x;
  long total = (long)kB * kNH * kN * kF;
  if (i >= total) return;
  int f = (int)(i % kF);
  long t = i / kF;
  int n = (int)(t % kN); t /= kN;
  int h = (int)(t % kNH);
  int b = (int)(t / kNH);
  float v = Hn[i];
  v = v > 0.f ? v : expm1f(v);
  H1[(((long)b * kN + n) * kNH + h) * kF + f] = v;
}

// wave-per-row layernorm; optional f32 and/or f16 outputs
__global__ __launch_bounds__(256)
void layernorm_kernel(const float* __restrict__ X, const float* __restrict__ g,
                      const float* __restrict__ bt, float* __restrict__ Yf,
                      f16* __restrict__ Yh, int Ddim, int rows) {
  int wid  = (blockIdx.x * blockDim.x + threadIdx.x) >> 5;
  int lane = threadIdx.x & 31;
  if (wid >= rows) return;
  const float* xr = X + (long)wid * Ddim;
  float v[16];
  float s = 0.f, ss = 0.f;
  int cnt = Ddim >> 5;                       // 16 for D=512, 2 for D=64
  for (int t = 0; t < cnt; ++t) {
    float x = xr[lane + 32 * t];
    v[t] = x; s += x; ss += x * x;
  }
  for (int off = 16; off; off >>= 1) {
    s  += __shfl_xor(s,  off, 32);
    ss += __shfl_xor(ss, off, 32);
  }
  float mu  = s / Ddim;
  float var = ss / Ddim - mu * mu;
  float r   = rsqrtf(var + kEPS);
  for (int t = 0; t < cnt; ++t) {
    int j = lane + 32 * t;
    float y = (v[t] - mu) * r * g[j] + bt[j];
    if (Yf) Yf[(long)wid * Ddim + j] = y;
    if (Yh) Yh[(long)wid * Ddim + j] = (f16)y;
  }
}

// Hn2 (B,NH,N,F) -> H2raw (B,N,F): mean over heads of ELU
__global__ void meanelu_kernel(const float* __restrict__ Hn,
                               float* __restrict__ H2) {
  long i = (long)blockIdx.x * blockDim.x + threadIdx.x;
  long total = (long)kB * kN * kF;
  if (i >= total) return;
  int f = (int)(i % kF);
  long t = i / kF;
  int n = (int)(t % kN);
  int b = (int)(t / kN);
  float acc = 0.f;
  for (int h = 0; h < kNH; ++h) {
    float v = Hn[(((long)b * kNH + h) * kN + n) * kF + f];
    acc += (v > 0.f ? v : expm1f(v));
  }
  H2[i] = acc * (1.0f / kNH);
}

// per-batch attention pooling: logits -> softmax over N -> weighted sum
__global__ __launch_bounds__(512)
void pool_kernel(const float* __restrict__ H2, const float* __restrict__ pw,
                 const float* __restrict__ pb, float* __restrict__ z) {
  __shared__ float sl[kN];
  __shared__ float red;
  int b = blockIdx.x;
  int n = threadIdx.x;
  const float* hr = H2 + ((long)b * kN + n) * kF;
  float lg = pb[0];
  for (int f = 0; f < kF; ++f) lg += hr[f] * pw[f];
  sl[n] = lg;
  __syncthreads();
  if (n == 0) {
    float m = sl[0];
    for (int j = 1; j < kN; ++j) m = fmaxf(m, sl[j]);
    red = m;
  }
  __syncthreads();
  float p = __expf(sl[n] - red);
  __syncthreads();
  sl[n] = p;
  __syncthreads();
  if (n == 0) {
    float s = 0.f;
    for (int j = 0; j < kN; ++j) s += sl[j];
    red = s;
  }
  __syncthreads();
  float inv = 1.0f / red;
  if (n < kF) {
    float acc = 0.f;
    for (int j = 0; j < kN; ++j) acc += sl[j] * H2[((long)b * kN + j) * kF + n];
    z[(long)b * kF + n] = acc * inv;
  }
}

// ---------------------------------------------------------------------------
extern "C" void kernel_launch(void* const* d_in, const int* in_sizes, int n_in,
                              void* d_out, int out_size, void* d_ws, size_t ws_size,
                              hipStream_t stream) {
  (void)in_sizes; (void)n_in; (void)out_size; (void)ws_size;
  const float* H     = (const float*)d_in[0];
  const float* W1    = (const float*)d_in[1];
  const float* a1    = (const float*)d_in[2];
  const float* W2    = (const float*)d_in[3];
  const float* a2    = (const float*)d_in[4];
  const float* ln1g  = (const float*)d_in[5];
  const float* ln1b  = (const float*)d_in[6];
  const float* ln2g  = (const float*)d_in[7];
  const float* ln2b  = (const float*)d_in[8];
  const float* poolw = (const float*)d_in[9];
  const float* poolb = (const float*)d_in[10];

  float* z      = (float*)d_out;                         // (16,64)
  float* alpha1 = z + (long)kB * kF;                     // (16,8,512,512)
  float* alpha2 = alpha1 + (long)kB * kNH * kN * kN;     // (16,8,512,512)

  // workspace carving (256B aligned); big0/big1 are reused across stages
  char* ws = (char*)d_ws;
  size_t off = 0;
  auto alloc = [&](size_t bytes) -> void* {
    void* p = ws + off;
    off = (off + bytes + 255) & ~(size_t)255;
    return p;
  };
  f16*   Hh    = (f16*)  alloc((size_t)kB * kN * kD * 2);        // H in f16
  f16*   W1t   = (f16*)  alloc((size_t)kNH * kD * kF * 2);       // W1^T f16
  float* big0  = (float*)alloc((size_t)kB * kNH * kN * kF * 4);  // Ht1/Hn1/Hn2
  f16*   Ht1t  = (f16*)  alloc((size_t)kB * kNH * kN * kF * 2);  // Ht1^T f16
  float* s1s   = (float*)alloc((size_t)kB * kNH * kN * 4);
  float* s1d   = (float*)alloc((size_t)kB * kNH * kN * 4);
  float* big1  = (float*)alloc((size_t)kB * kN * kNH * kF * 4);  // H1raw/Ht2
  f16*   H1h   = (f16*)  alloc((size_t)kB * kN * kNH * kF * 2);  // LN1 out f16
  f16*   W2t   = (f16*)  alloc((size_t)kNH * kNH * kF * kF * 2); // W2^T f16
  f16*   Ht2t  = (f16*)  alloc((size_t)kB * kNH * kN * kF * 2);  // Ht2^T f16
  float* s2s   = (float*)alloc((size_t)kB * kNH * kN * 4);
  float* s2d   = (float*)alloc((size_t)kB * kNH * kN * 4);
  float* H2raw = (float*)alloc((size_t)kB * kN * kF * 4);
  float* H2    = (float*)alloc((size_t)kB * kN * kF * 4);

  const int rowsBHN = kB * kNH * kN;                     // 65536
  const int nb = kB * kNH;                               // 128
  const int gemmBlocks = nb * ((kN / 16) / 8);           // 512

  // ---- layer 1 -------------------------------------------------------------
  { long n = (long)kB * kN * kD;
    cast_f16_kernel<<<(int)((n + 255) / 256), 256, 0, stream>>>(H, Hh, n); }
  { long n = (long)kNH * kD * kF;
    tcast_kernel<<<(int)((n + 255) / 256), 256, 0, stream>>>(W1, W1t, kD, kF, kNH); }
  // Ht1 = H x W1 : per bh, (512x128)*(128x64); LDS panel = 64*128*2 = 16KB
  gemm_wmma<false><<<gemmBlocks, 256, 64 * kD * 2, stream>>>(
      Hh, W1t, big0, kN, kD, nb,
      (long)kN * kD, 0L, 0L, (long)kF * kD, (long)kN * kF);
  scores_kernel<<<(rowsBHN + 255) / 256, 256, 0, stream>>>(big0, a1, s1s, s1d);
  softmax_kernel<<<(int)(((long)rowsBHN * 32 + 255) / 256), 256, 0, stream>>>(
      s1s, s1d, alpha1);
  { long n = (long)kB * kNH * kN * kF;
    tcast_kernel<<<(int)((n + 255) / 256), 256, 0, stream>>>(
        big0, Ht1t, kN, kF, nb); }
  // Hn1 = alpha1 (f32) x Ht1 : per bh, (512x512)*(512x64); LDS panel 64KB
  gemm_wmma<true><<<gemmBlocks, 256, 64 * kN * 2, stream>>>(
      alpha1, Ht1t, big0, kN, kN, nb,
      (long)kNH * kN * kN, (long)kN * kN,
      (long)kNH * kF * kN, (long)kF * kN, (long)kN * kF);
  { long n = (long)kB * kNH * kN * kF;
    eluconcat_kernel<<<(int)((n + 255) / 256), 256, 0, stream>>>(big0, big1); }
  { int rows = kB * kN;
    layernorm_kernel<<<(int)(((long)rows * 32 + 255) / 256), 256, 0, stream>>>(
        big1, ln1g, ln1b, nullptr, H1h, kNH * kF, rows); }

  // ---- layer 2 -------------------------------------------------------------
  { long n = (long)kNH * kNH * kF * kF;
    tcast_kernel<<<(int)((n + 255) / 256), 256, 0, stream>>>(
        W2, W2t, kNH * kF, kF, kNH); }
  // Ht2 = H1 x W2 : per bh, (512x512)*(512x64); overwrites H1raw (big1)
  gemm_wmma<false><<<gemmBlocks, 256, 64 * (kNH * kF) * 2, stream>>>(
      H1h, W2t, big1, kN, kNH * kF, nb,
      (long)kN * kNH * kF, 0L, 0L, (long)kF * kNH * kF, (long)kN * kF);
  scores_kernel<<<(rowsBHN + 255) / 256, 256, 0, stream>>>(big1, a2, s2s, s2d);
  softmax_kernel<<<(int)(((long)rowsBHN * 32 + 255) / 256), 256, 0, stream>>>(
      s2s, s2d, alpha2);
  { long n = (long)kB * kNH * kN * kF;
    tcast_kernel<<<(int)((n + 255) / 256), 256, 0, stream>>>(
        big1, Ht2t, kN, kF, nb); }
  // Hn2 = alpha2 (f32) x Ht2
  gemm_wmma<true><<<gemmBlocks, 256, 64 * kN * 2, stream>>>(
      alpha2, Ht2t, big0, kN, kN, nb,
      (long)kNH * kN * kN, (long)kN * kN,
      (long)kNH * kF * kN, (long)kF * kN, (long)kN * kF);
  { long n = (long)kB * kN * kF;
    meanelu_kernel<<<(int)((n + 255) / 256), 256, 0, stream>>>(big0, H2raw); }
  { int rows = kB * kN;
    layernorm_kernel<<<(int)(((long)rows * 32 + 255) / 256), 256, 0, stream>>>(
        H2raw, ln2g, ln2b, H2, nullptr, kF, rows); }
  pool_kernel<<<kB, kN, 0, stream>>>(H2, poolw, poolb, z);
}
